// MTREncoder_16930761080939
// MI455X (gfx1250) — compile-verified, matrix operand order checked by reading, not run on hardware
//
#include <hip/hip_runtime.h>
#include <hip/hip_bf16.h>
#include <math.h>

typedef __attribute__((ext_vector_type(16))) _Float16 v16h;
typedef __attribute__((ext_vector_type(8)))  _Float16 v8h;
typedef __attribute__((ext_vector_type(8)))  float    v8f;
typedef __attribute__((ext_vector_type(4)))  float    v4f;
typedef __attribute__((ext_vector_type(4)))  unsigned int v4u;

#define CDIV(a,b) (((a)+(b)-1)/(b))

namespace {
constexpr int B_   = 8,  NO = 128, T_ = 11, CAc = 29;
constexpr int NPp  = 512, PPTS = 20, CMc = 9;
constexpr int Dd   = 256, Hh = 8,  Kk = 16, Ll = 6, FFd = 1024;
constexpr int NTOK = NO + NPp;        // 640
constexpr int ROWS = B_ * NTOK;       // 5120
constexpr int APTS = B_ * NO * T_;    // 11264 agent points
constexpr int MPTS = B_ * NPp * PPTS; // 81920 map points
constexpr int APLY = B_ * NO;         // 1024 agent polylines
constexpr int MPLY = B_ * NPp;        // 4096 map polylines

constexpr int BM = 64, BN = 64, BK = 32;   // block tile
}

__device__ __forceinline__ int imin(int a, int b) { return a < b ? a : b; }

union FragU { v16h h; v4u q[2]; };
union PairU { _Float16 h[2]; unsigned int u; };

// ---------------------------------------------------------------------------
// WMMA GEMM: Out[M,N] = act( A[M,K] (+Aadd) @ W[K,N] + bias (+resid) )
// 256 threads = 8 waves; block computes 64x64; wave computes 16x32 (2 WMMAs
// per K-chunk). A and B staged in LDS as f16, B stored K-major so fragments
// are two contiguous ds_load_b128. Block-uniform fast path: vectorized b128
// global loads, packed f16 conversion, B transpose via paired ds_store_b32.
// Edge path (only the K=30 / K=9 GEMMs) uses branchless clamped scalars.
// ---------------------------------------------------------------------------
template<bool ADD, bool RES, bool RELU>
__global__ __launch_bounds__(256)
void wmma_gemm(const float* __restrict__ A, const float* __restrict__ Aadd,
               const float* __restrict__ W, const float* __restrict__ bias,
               const float* __restrict__ resid, float* __restrict__ Out,
               int M, int N, int Kd)
{
    __shared__ _Float16 lA[BM * BK];   // [m][k], row stride BK
    __shared__ _Float16 lB[BN * BK];   // [n][k] (K-major), row stride BK

    const int t    = threadIdx.x;
    const int lane = t & 31;
    const int wave = t >> 5;
    const int wm   = wave >> 1;          // 0..3  -> m subtile
    const int wn   = wave & 1;           // 0..1  -> n subtile (32 wide)
    const int m0   = blockIdx.x * BM;
    const int n0   = blockIdx.y * BN;
    const int r    = lane & 15;
    const int hf   = lane >> 4;

    // A staging coords (8 contiguous K per thread)
    const int arow = t >> 2;             // 0..63
    const int aseg = (t & 3) * 8;        // 0,8,16,24
    const int gm   = m0 + arow;
    const int mm   = imin(gm, M - 1);
    // B edge-path coords (K-major scalar)
    const int gn   = n0 + arow;
    const int nn   = imin(gn, N - 1);
    // B fast-path coords: 4-wide n slice of two adjacent k rows
    const int bn4  = (t >> 4) * 4;       // 0..60  (n within tile)
    const int bkp  = (t & 15) * 2;       // 0..30  (even k within chunk)

    const bool mnFull = (m0 + BM <= M) && (n0 + BN <= N);

    v8f acc0 = {}, acc1 = {};

    for (int k0 = 0; k0 < Kd; k0 += BK) {
        // prefetch next A chunk (uniform condition; emits global_prefetch_b8)
        if (k0 + BK < Kd)
            __builtin_prefetch(A + (size_t)mm * Kd + k0 + BK, 0, 1);

        if (mnFull && (k0 + BK <= Kd)) {
            // ---------- fast path: fully in-bounds, vector loads ----------
            const float* pA = A + (size_t)gm * Kd + (k0 + aseg);
            v4f x0 = *(const v4f*)pA;
            v4f x1 = *(const v4f*)(pA + 4);
            if (ADD) {
                const float* pD = Aadd + (size_t)gm * Kd + (k0 + aseg);
                v4f d0 = *(const v4f*)pD;
                v4f d1 = *(const v4f*)(pD + 4);
                x0 += d0;
                x1 += d1;
            }
            v8h pk;
#pragma unroll
            for (int j = 0; j < 4; ++j) {
                pk[j]     = (_Float16)x0[j];
                pk[4 + j] = (_Float16)x1[j];
            }
            *(v8h*)&lA[arow * BK + aseg] = pk;

            // B: two adjacent k rows, 4 n each; transpose-store as packed b32
            const float* pB0 = W + (size_t)(k0 + bkp) * N + (n0 + bn4);
            const float* pB1 = pB0 + N;
            v4f y0 = *(const v4f*)pB0;
            v4f y1 = *(const v4f*)pB1;
#pragma unroll
            for (int j = 0; j < 4; ++j) {
                PairU pp;
                pp.h[0] = (_Float16)y0[j];   // k = bkp
                pp.h[1] = (_Float16)y1[j];   // k = bkp+1
                *(unsigned int*)&lB[(bn4 + j) * BK + bkp] = pp.u;
            }
        } else {
            // ---------- edge path: clamped scalar loads, branchless ----------
            {
                v8h pk;
#pragma unroll
                for (int j = 0; j < 8; ++j) {
                    const int k  = k0 + aseg + j;
                    const int kc = imin(k, Kd - 1);
                    float x = A[(size_t)mm * Kd + kc];
                    if (ADD) x += Aadd[(size_t)mm * Kd + kc];
                    x = (gm < M && k < Kd) ? x : 0.f;
                    pk[j] = (_Float16)x;
                }
                *(v8h*)&lA[arow * BK + aseg] = pk;
            }
            {
                v8h pk;
#pragma unroll
                for (int j = 0; j < 8; ++j) {
                    const int k  = k0 + aseg + j;
                    const int kc = imin(k, Kd - 1);
                    float x = W[(size_t)kc * N + nn];
                    x = (gn < N && k < Kd) ? x : 0.f;
                    pk[j] = (_Float16)x;
                }
                *(v8h*)&lB[arow * BK + aseg] = pk;
            }
        }
        __syncthreads();

        // ---- fragments: two ds_load_b128 each ----
        FragU af, bf0, bf1;
        {
            const v4u* pa = (const v4u*)&lA[(wm * 16 + r) * BK];
            af.q[0] = pa[hf];
            af.q[1] = pa[2 + hf];
            const v4u* pb0 = (const v4u*)&lB[(wn * 32 + r) * BK];
            bf0.q[0] = pb0[hf];
            bf0.q[1] = pb0[2 + hf];
            const v4u* pb1 = (const v4u*)&lB[(wn * 32 + 16 + r) * BK];
            bf1.q[0] = pb1[hf];
            bf1.q[1] = pb1[2 + hf];
        }

        acc0 = __builtin_amdgcn_wmma_f32_16x16x32_f16(
            false, af.h, false, bf0.h, (short)0, acc0, false, false);
        acc1 = __builtin_amdgcn_wmma_f32_16x16x32_f16(
            false, af.h, false, bf1.h, (short)0, acc1, false, false);

        __syncthreads();
    }

    // ---- epilogue ----
    const int em = m0 + wm * 16;
    const int en = n0 + wn * 32;
#pragma unroll
    for (int v = 0; v < 8; ++v) {
        const int m = em + v + 8 * hf;
        if (m < M) {
            const int na = en + r;
            if (na < N) {
                float o = acc0[v] + bias[na];
                if (RES)  o += resid[(size_t)m * N + na];
                if (RELU) o = fmaxf(o, 0.f);
                Out[(size_t)m * N + na] = o;
            }
            const int nb = en + 16 + r;
            if (nb < N) {
                float o = acc1[v] + bias[nb];
                if (RES)  o += resid[(size_t)m * N + nb];
                if (RELU) o = fmaxf(o, 0.f);
                Out[(size_t)m * N + nb] = o;
            }
        }
    }
}

// ---------------------------------------------------------------------------
// Elementwise / reduction helpers
// ---------------------------------------------------------------------------
__global__ void obj_in_kernel(const float* __restrict__ obj, float* __restrict__ out)
{
    const int total = APTS * (CAc + 1);
    int i = blockIdx.x * blockDim.x + threadIdx.x;
    if (i >= total) return;
    const int c = i % (CAc + 1);
    const int p = i / (CAc + 1);
    out[i] = (c < CAc) ? obj[(size_t)p * CAc + c] : 1.0f;  // mask channel == 1
}

__global__ void maxpool_kernel(const float* __restrict__ in, float* __restrict__ out,
                               int P, int T, int C)
{
    int i = blockIdx.x * blockDim.x + threadIdx.x;
    if (i >= P * C) return;
    const int c = i % C, p = i / C;
    float m = -3.4e38f;
    for (int t = 0; t < T; ++t)
        m = fmaxf(m, in[((size_t)p * T + t) * C + c]);
    out[i] = m;
}

__global__ void catpool_kernel(const float* __restrict__ x, const float* __restrict__ pooled,
                               float* __restrict__ out, int P, int T, int C)
{
    const long total = (long)P * T * 2 * C;
    long i = (long)blockIdx.x * blockDim.x + threadIdx.x;
    if (i >= total) return;
    const int  c  = (int)(i % (2 * C));
    const long pt = i / (2 * C);
    const long p  = pt / T;
    out[i] = (c < C) ? x[pt * C + c] : pooled[p * C + (c - C)];
}

__global__ void cat_tokens_kernel(const float* __restrict__ objF, const float* __restrict__ mapF,
                                  float* __restrict__ x)
{
    int i = blockIdx.x * blockDim.x + threadIdx.x;
    if (i >= ROWS * Dd) return;
    const int c = i % Dd, tok = i / Dd;
    const int b = tok / NTOK, n = tok % NTOK;
    x[i] = (n < NO) ? objF[((size_t)b * NO + n) * Dd + c]
                    : mapF[((size_t)b * NPp + (n - NO)) * Dd + c];
}

__global__ void cat_pos_kernel(const float* __restrict__ op, const float* __restrict__ mp,
                               float* __restrict__ pos)
{
    int i = blockIdx.x * blockDim.x + threadIdx.x;
    if (i >= ROWS * 3) return;
    const int c = i % 3, tok = i / 3;
    const int b = tok / NTOK, n = tok % NTOK;
    pos[i] = (n < NO) ? op[((size_t)b * NO + n) * 3 + c]
                      : mp[((size_t)b * NPp + (n - NO)) * 3 + c];
}

__global__ void knn_kernel(const float* __restrict__ pos, int* __restrict__ nbr)
{
    int t = blockIdx.x * blockDim.x + threadIdx.x;
    if (t >= ROWS) return;
    const int b = t / NTOK;
    const float* pb = pos + (size_t)b * NTOK * 3;
    const float px = pos[t * 3], py = pos[t * 3 + 1], pz = pos[t * 3 + 2];
    float bd[Kk]; int bi[Kk];
#pragma unroll
    for (int k = 0; k < Kk; ++k) { bd[k] = 3.4e38f; bi[k] = 0; }
    for (int j = 0; j < NTOK; ++j) {
        const float dx = pb[j * 3] - px, dy = pb[j * 3 + 1] - py, dz = pb[j * 3 + 2] - pz;
        const float d2 = dx * dx + dy * dy + dz * dz;
        if (d2 < bd[Kk - 1]) {
            int p = Kk - 1;
            while (p > 0 && d2 < bd[p - 1]) { bd[p] = bd[p - 1]; bi[p] = bi[p - 1]; --p; }
            bd[p] = d2; bi[p] = j;
        }
    }
#pragma unroll
    for (int k = 0; k < Kk; ++k) nbr[(size_t)t * Kk + k] = bi[k];
}

__global__ void pe_kernel(const float* __restrict__ pos, float* __restrict__ pe)
{
    int i = blockIdx.x * blockDim.x + threadIdx.x;
    if (i >= ROWS * Dd) return;
    const int c = i % Dd, tok = i / Dd;
    const int half = Dd / 2;  // 128
    int j; float coord;
    if (c < half) { j = c;        coord = pos[tok * 3 + 1]; }  // y first
    else          { j = c - half; coord = pos[tok * 3 + 0]; }  // then x
    const float dimt = powf(10000.f, (float)(j & ~1) / (float)half);
    const float v = coord * 6.283185307179586f / dimt;
    pe[i] = (j & 1) ? cosf(v) : sinf(v);
}

__global__ void attn_kernel(const float* __restrict__ q, const float* __restrict__ k,
                            const float* __restrict__ v, const int* __restrict__ nbr,
                            float* __restrict__ o)
{
    int t = blockIdx.x * blockDim.x + threadIdx.x;
    if (t >= ROWS * Hh) return;
    const int h = t % Hh, tok = t / Hh, b = tok / NTOK;
    const int hd = Dd / Hh;  // 32
    float qr[32], acc[32];
    const float* qp = q + (size_t)tok * Dd + h * hd;
#pragma unroll
    for (int d = 0; d < 32; ++d) { qr[d] = qp[d]; acc[d] = 0.f; }
    float s[Kk]; float mx = -3.4e38f;
    for (int kk = 0; kk < Kk; ++kk) {
        const int idx = nbr[(size_t)tok * Kk + kk];
        const float* kp = k + ((size_t)b * NTOK + idx) * Dd + h * hd;
        float dot = 0.f;
#pragma unroll
        for (int d = 0; d < 32; ++d) dot += qr[d] * kp[d];
        s[kk] = dot * 0.17677669529663687f;  // 1/sqrt(32)
        mx = fmaxf(mx, s[kk]);
    }
    float den = 0.f;
    for (int kk = 0; kk < Kk; ++kk) { s[kk] = expf(s[kk] - mx); den += s[kk]; }
    const float inv = 1.f / den;
    for (int kk = 0; kk < Kk; ++kk) {
        const int idx = nbr[(size_t)tok * Kk + kk];
        const float* vp = v + ((size_t)b * NTOK + idx) * Dd + h * hd;
        const float w = s[kk] * inv;
#pragma unroll
        for (int d = 0; d < 32; ++d) acc[d] += w * vp[d];
    }
    float* op = o + (size_t)tok * Dd + h * hd;
#pragma unroll
    for (int d = 0; d < 32; ++d) op[d] = acc[d];
}

__global__ void ln_kernel(const float* __restrict__ in, const float* __restrict__ g,
                          const float* __restrict__ b, float* __restrict__ out,
                          int rows, int C)
{
    int r = blockIdx.x * blockDim.x + threadIdx.x;
    if (r >= rows) return;
    const float* x = in + (size_t)r * C;
    float mu = 0.f;
    for (int c = 0; c < C; ++c) mu += x[c];
    mu /= (float)C;
    float var = 0.f;
    for (int c = 0; c < C; ++c) { const float d = x[c] - mu; var += d * d; }
    var /= (float)C;
    const float inv = rsqrtf(var + 1e-5f);
    float* y = out + (size_t)r * C;
    for (int c = 0; c < C; ++c) y[c] = (x[c] - mu) * inv * g[c] + b[c];
}

// ---------------------------------------------------------------------------
// Host launch
// ---------------------------------------------------------------------------
static inline void gemm(hipStream_t s, const float* A, const float* Aadd, const float* W,
                        const float* bias, const float* resid, float* Out,
                        int M, int N, int K, int relu)
{
    dim3 grid(CDIV(M, BM), CDIV(N, BN));
    if (Aadd)
        wmma_gemm<true, false, false><<<grid, 256, 0, s>>>(A, Aadd, W, bias, resid, Out, M, N, K);
    else if (resid)
        wmma_gemm<false, true, false><<<grid, 256, 0, s>>>(A, Aadd, W, bias, resid, Out, M, N, K);
    else if (relu)
        wmma_gemm<false, false, true><<<grid, 256, 0, s>>>(A, Aadd, W, bias, resid, Out, M, N, K);
    else
        wmma_gemm<false, false, false><<<grid, 256, 0, s>>>(A, Aadd, W, bias, resid, Out, M, N, K);
}

extern "C" void kernel_launch(void* const* d_in, const int* in_sizes, int n_in,
                              void* d_out, int out_size, void* d_ws, size_t ws_size,
                              hipStream_t stream)
{
    (void)in_sizes; (void)n_in; (void)out_size; (void)ws_size;

    const float* obj_trajs = (const float*)d_in[0];
    const float* map_poly  = (const float*)d_in[1];
    const float* obj_pos   = (const float*)d_in[2];
    const float* map_cent  = (const float*)d_in[3];

    auto PF = [&](int i) { return (const float*)d_in[i]; };
    // params flattened in JAX pytree order (sorted dict keys): agent, layers, map
    const int P0 = 4;
    const float *agW_mid0 = PF(P0 + 0), *agB_mid0 = PF(P0 + 1);
    const float *agW_mid1 = PF(P0 + 2), *agB_mid1 = PF(P0 + 3);
    const float *agW_out0 = PF(P0 + 4), *agB_out0 = PF(P0 + 5);
    const float *agW_out1 = PF(P0 + 6), *agB_out1 = PF(P0 + 7);
    const float *agW_pre0 = PF(P0 + 8), *agB_pre0 = PF(P0 + 9);
    const int LB = P0 + 10;           // 6 layers x 16 sorted leaves
    const int MB = LB + 16 * Ll;
    const float *mpW_mid0 = PF(MB + 0),  *mpB_mid0 = PF(MB + 1);
    const float *mpW_mid1 = PF(MB + 2),  *mpB_mid1 = PF(MB + 3);
    const float *mpW_out0 = PF(MB + 4),  *mpB_out0 = PF(MB + 5);
    const float *mpW_out1 = PF(MB + 6),  *mpB_out1 = PF(MB + 7);
    const float *mpW_pre0 = PF(MB + 8),  *mpB_pre0 = PF(MB + 9);
    const float *mpW_pre1 = PF(MB + 10), *mpB_pre1 = PF(MB + 11);
    const float *mpW_pre2 = PF(MB + 12), *mpB_pre2 = PF(MB + 13);

    // ---- workspace carving ----
    float* ws = (float*)d_ws;
    size_t off = 0;
    auto alloc = [&](size_t n) { float* p = ws + off; off += n; return p; };
    float* objF = alloc((size_t)APLY * Dd);
    float* mapF = alloc((size_t)MPLY * Dd);
    float* xbuf = alloc((size_t)ROWS * Dd);
    float* pe   = alloc((size_t)ROWS * Dd);
    float* pos  = alloc((size_t)ROWS * 3);
    int*   nbr  = (int*)alloc((size_t)ROWS * Kk);
    float* x2   = alloc((size_t)ROWS * Dd);
    float* qb   = alloc((size_t)ROWS * Dd);
    float* kb   = alloc((size_t)ROWS * Dd);
    float* vb   = alloc((size_t)ROWS * Dd);
    float* ob   = alloc((size_t)ROWS * Dd);
    float* tmp  = alloc((size_t)ROWS * Dd);
    float* scratch = ws + off;  // reused across phases

    // ================= agent PointNet =================
    {
        float* objin = scratch;                             // APTS x 30
        float* aA    = objin + (size_t)APTS * 30;           // APTS x 256
        float* aPool = aA    + (size_t)APTS * 256;          // APLY x 256
        float* aCat  = aPool + (size_t)APLY * 256;          // APTS x 512
        float* aB    = aCat  + (size_t)APTS * 512;          // APTS x 256
        float* aH    = aB    + (size_t)APTS * 256;          // APLY x 256

        const int n1 = APTS * 30;
        obj_in_kernel<<<CDIV(n1, 256), 256, 0, stream>>>(obj_trajs, objin);
        gemm(stream, objin, nullptr, agW_pre0, agB_pre0, nullptr, aA, APTS, 256, 30, 1);
        maxpool_kernel<<<CDIV(APLY * 256, 256), 256, 0, stream>>>(aA, aPool, APLY, T_, 256);
        catpool_kernel<<<CDIV(APTS * 512, 256), 256, 0, stream>>>(aA, aPool, aCat, APLY, T_, 256);
        gemm(stream, aCat, nullptr, agW_mid0, agB_mid0, nullptr, aB, APTS, 256, 512, 1);
        gemm(stream, aB,   nullptr, agW_mid1, agB_mid1, nullptr, aA, APTS, 256, 256, 1);
        maxpool_kernel<<<CDIV(APLY * 256, 256), 256, 0, stream>>>(aA, aPool, APLY, T_, 256);
        gemm(stream, aPool, nullptr, agW_out0, agB_out0, nullptr, aH,   APLY, 256, 256, 1);
        gemm(stream, aH,    nullptr, agW_out1, agB_out1, nullptr, objF, APLY, 256, 256, 0);
    }

    // ================= map PointNet =================
    {
        float* mA    = scratch;                              // MPTS x 64
        float* mB    = mA    + (size_t)MPTS * 64;
        float* mC    = mB    + (size_t)MPTS * 64;
        float* mPool = mC    + (size_t)MPTS * 64;            // MPLY x 64
        float* mCat  = mPool + (size_t)MPLY * 64;            // MPTS x 128
        float* mH    = mCat  + (size_t)MPTS * 128;           // MPLY x 64

        gemm(stream, map_poly, nullptr, mpW_pre0, mpB_pre0, nullptr, mA, MPTS, 64, 9,  1);
        gemm(stream, mA,       nullptr, mpW_pre1, mpB_pre1, nullptr, mB, MPTS, 64, 64, 1);
        gemm(stream, mB,       nullptr, mpW_pre2, mpB_pre2, nullptr, mC, MPTS, 64, 64, 1);
        maxpool_kernel<<<CDIV(MPLY * 64, 256), 256, 0, stream>>>(mC, mPool, MPLY, PPTS, 64);
        catpool_kernel<<<CDIV(MPTS * 128, 256), 256, 0, stream>>>(mC, mPool, mCat, MPLY, PPTS, 64);
        gemm(stream, mCat, nullptr, mpW_mid0, mpB_mid0, nullptr, mA, MPTS, 64, 128, 1);
        gemm(stream, mA,   nullptr, mpW_mid1, mpB_mid1, nullptr, mB, MPTS, 64, 64,  1);
        maxpool_kernel<<<CDIV(MPLY * 64, 256), 256, 0, stream>>>(mB, mPool, MPLY, PPTS, 64);
        gemm(stream, mPool, nullptr, mpW_out0, mpB_out0, nullptr, mH,   MPLY, 64,  64, 1);
        gemm(stream, mH,    nullptr, mpW_out1, mpB_out1, nullptr, mapF, MPLY, 256, 64, 0);
    }

    // ================= token assembly =================
    cat_tokens_kernel<<<CDIV(ROWS * Dd, 256), 256, 0, stream>>>(objF, mapF, xbuf);
    cat_pos_kernel<<<CDIV(ROWS * 3, 256), 256, 0, stream>>>(obj_pos, map_cent, pos);
    knn_kernel<<<CDIV(ROWS, 128), 128, 0, stream>>>(pos, nbr);
    pe_kernel<<<CDIV(ROWS * Dd, 256), 256, 0, stream>>>(pos, pe);

    // ================= transformer layers =================
    float* ffn = scratch;  // ROWS x FF, reuses PointNet scratch
    for (int l = 0; l < Ll; ++l) {
        const int base = LB + l * 16;
        const float *W1 = PF(base + 0),  *W2 = PF(base + 1);
        const float *Wk = PF(base + 2),  *Wo = PF(base + 3);
        const float *Wq = PF(base + 4),  *Wv = PF(base + 5);
        const float *b1 = PF(base + 6),  *b2 = PF(base + 7);
        const float *be1 = PF(base + 8), *be2 = PF(base + 9);
        const float *bk = PF(base + 10), *bo = PF(base + 11);
        const float *bq = PF(base + 12), *bv = PF(base + 13);
        const float *g1 = PF(base + 14), *g2 = PF(base + 15);

        gemm(stream, xbuf, pe,      Wq, bq, nullptr, qb, ROWS, Dd, Dd, 0);
        gemm(stream, xbuf, pe,      Wk, bk, nullptr, kb, ROWS, Dd, Dd, 0);
        gemm(stream, xbuf, nullptr, Wv, bv, nullptr, vb, ROWS, Dd, Dd, 0);
        attn_kernel<<<CDIV(ROWS * Hh, 128), 128, 0, stream>>>(qb, kb, vb, nbr, ob);
        gemm(stream, ob, nullptr, Wo, bo, xbuf, tmp, ROWS, Dd, Dd, 0);
        ln_kernel<<<CDIV(ROWS, 128), 128, 0, stream>>>(tmp, g1, be1, x2, ROWS, Dd);
        gemm(stream, x2,  nullptr, W1, b1, nullptr, ffn, ROWS, FFd, Dd,  1);
        gemm(stream, ffn, nullptr, W2, b2, x2,      tmp, ROWS, Dd,  FFd, 0);
        float* out = (l == Ll - 1) ? (float*)d_out : xbuf;
        ln_kernel<<<CDIV(ROWS, 128), 128, 0, stream>>>(tmp, g2, be2, out, ROWS, Dd);
    }
}
